// MyGCN_55353538510962
// MI455X (gfx1250) — compile-verified, matrix operand order checked by reading, not run on hardware
//
#include <hip/hip_runtime.h>

typedef __attribute__((ext_vector_type(16))) _Float16 v16h;
typedef __attribute__((ext_vector_type(8)))  _Float16 v8h;
typedef __attribute__((ext_vector_type(8)))  float    v8f;

#define IN_SZ  128
#define HID_SZ 128
#define OUT_SZ 64

__device__ __forceinline__ void atomicAddF32(float* p, float v) {
#if defined(__HIP_DEVICE_COMPILE__)
  unsafeAtomicAdd(p, v);   // global_atomic_add_f32
#else
  atomicAdd(p, v);
#endif
}

// ---------------- utility kernels ----------------

__global__ void fill_f32(float* p, float v, long long n) {
  long long i = (long long)blockIdx.x * blockDim.x + threadIdx.x;
  long long stride = (long long)gridDim.x * blockDim.x;
  for (; i < n; i += stride) p[i] = v;
}

// Pre-swizzle a KxN f32 weight matrix into WMMA B-fragment order (f16):
// ws[((nt*ktiles + kt)*32 + lane)*16 + i] = (f16) w[k*N + col]
// where lh=lane&15, hi8=lane>>4, k0=kt*32+hi8*8,
//       k = i<8 ? k0+i : k0+16+(i-8), col = nt*16+lh.
// Each lane's 16 halves are then one contiguous 32B fragment load in the GEMM.
__global__ void swizzle_w(const float* w, _Float16* ws, int N, int ktiles) {
  int idx = blockIdx.x * blockDim.x + threadIdx.x;
  int total = (N / 16) * ktiles * 32 * 16;
  if (idx >= total) return;
  int i    = idx & 15;
  int lane = (idx >> 4) & 31;
  int kt   = (idx >> 9) % ktiles;
  int nt   = (idx >> 9) / ktiles;
  int lh = lane & 15, hi8 = lane >> 4;
  int k0 = kt * 32 + hi8 * 8;
  int k  = (i < 8) ? (k0 + i) : (k0 + 16 + (i - 8));
  int col = nt * 16 + lh;
  ws[idx] = (_Float16)w[k * N + col];
}

// degree: deg pre-filled with 1.0 (self loop); count in-edges on dst
__global__ void deg_count(const long long* dst_idx, float* deg, int n_edges) {
  int i = blockIdx.x * blockDim.x + threadIdx.x;
  int stride = gridDim.x * blockDim.x;
  for (; i < n_edges; i += stride)
    atomicAddF32(&deg[(int)dst_idx[i]], 1.0f);
}

__global__ void rs_kernel(const float* deg, float* rs, int n) {
  int i = blockIdx.x * blockDim.x + threadIdx.x;
  if (i < n) rs[i] = rsqrtf(deg[i]);
}

// ---------------- edge aggregation (wave per edge) ----------------
// acc[dst] += rs[src] * x[src]   (128 features, float4 per lane)
__global__ __launch_bounds__(256)
void agg_edges_128(const long long* src_idx, const long long* dst_idx,
                   const float* rs, const float* x, float* acc, int n_edges) {
  long long tid = (long long)blockIdx.x * blockDim.x + threadIdx.x;
  int e    = (int)(tid >> 5);
  int lane = (int)(tid & 31);
  if (e >= n_edges) return;
  int s = (int)src_idx[e];
  int d = (int)dst_idx[e];
  int ep = e + 2048;
  if (ep < n_edges) {
    int sp = (int)src_idx[ep];
    __builtin_prefetch(&x[(size_t)sp * IN_SZ + lane * 4], 0, 1);
  }
  float rsv = rs[s];
  const float4* xr = (const float4*)(x + (size_t)s * IN_SZ);
  float4 v = xr[lane];
  float* ar = acc + (size_t)d * IN_SZ + lane * 4;
  atomicAddF32(ar + 0, rsv * v.x);
  atomicAddF32(ar + 1, rsv * v.y);
  atomicAddF32(ar + 2, rsv * v.z);
  atomicAddF32(ar + 3, rsv * v.w);
}

// acc[dst] += ps[src]   (64 features, float2 per lane; ps already rs-scaled)
__global__ __launch_bounds__(256)
void agg_edges_64(const long long* src_idx, const long long* dst_idx,
                  const float* ps, float* acc, int n_edges) {
  long long tid = (long long)blockIdx.x * blockDim.x + threadIdx.x;
  int e    = (int)(tid >> 5);
  int lane = (int)(tid & 31);
  if (e >= n_edges) return;
  int s = (int)src_idx[e];
  int d = (int)dst_idx[e];
  int ep = e + 2048;
  if (ep < n_edges) {
    int sp = (int)src_idx[ep];
    __builtin_prefetch(&ps[(size_t)sp * OUT_SZ + lane * 2], 0, 1);
  }
  const float2* pr = (const float2*)(ps + (size_t)s * OUT_SZ);
  float2 v = pr[lane];
  float* ar = acc + (size_t)d * OUT_SZ + lane * 2;
  atomicAddF32(ar + 0, v.x);
  atomicAddF32(ar + 1, v.y);
}

// ---------------- WMMA GEMM layer 1 ----------------
// A[row][k] = rs[row]*(acc1[row][k] + rs[row]*x[row][k])  (self-loop + norm folded)
// h = relu(A @ w1 + b1), stored f16.
// grid.x = n_nodes/16 row tiles; block = 256 = 8 waves; wave w -> N-tile w.
__global__ __launch_bounds__(256)
void gemm1_wmma(const float* acc1, const float* x, const float* rs,
                const v16h* w1s, const float* b1, _Float16* h, int n_nodes) {
  int wave = threadIdx.x >> 5;
  int lane = threadIdx.x & 31;
  int lh   = lane & 15;
  int hi8  = lane >> 4;
  int mt   = blockIdx.x;
  int nt   = wave;
  int row  = mt * 16 + lh;
  bool rvalid = row < n_nodes;
  int arow = rvalid ? row : 0;
  float rsv = rvalid ? rs[arow] : 0.0f;
  const float* pa = acc1 + (size_t)arow * IN_SZ;
  const float* px = x    + (size_t)arow * IN_SZ;
  int col = nt * 16 + lh;

  v8f c = {0.f, 0.f, 0.f, 0.f, 0.f, 0.f, 0.f, 0.f};
#pragma unroll
  for (int kt = 0; kt < 4; ++kt) {
    int k0 = kt * 32 + hi8 * 8;
    // one contiguous 32B fragment per lane (pre-swizzled)
    v16h b = w1s[(nt * 4 + kt) * 32 + lane];
    // A: vector loads of 8+8 consecutive f32, fuse self-loop + norm, cvt to f16
    float4 a0 = *(const float4*)(pa + k0);
    float4 a1 = *(const float4*)(pa + k0 + 4);
    float4 a2 = *(const float4*)(pa + k0 + 16);
    float4 a3 = *(const float4*)(pa + k0 + 20);
    float4 x0 = *(const float4*)(px + k0);
    float4 x1 = *(const float4*)(px + k0 + 4);
    float4 x2 = *(const float4*)(px + k0 + 16);
    float4 x3 = *(const float4*)(px + k0 + 20);
    v16h a;
    a[0]  = (_Float16)(rsv * (a0.x + rsv * x0.x));
    a[1]  = (_Float16)(rsv * (a0.y + rsv * x0.y));
    a[2]  = (_Float16)(rsv * (a0.z + rsv * x0.z));
    a[3]  = (_Float16)(rsv * (a0.w + rsv * x0.w));
    a[4]  = (_Float16)(rsv * (a1.x + rsv * x1.x));
    a[5]  = (_Float16)(rsv * (a1.y + rsv * x1.y));
    a[6]  = (_Float16)(rsv * (a1.z + rsv * x1.z));
    a[7]  = (_Float16)(rsv * (a1.w + rsv * x1.w));
    a[8]  = (_Float16)(rsv * (a2.x + rsv * x2.x));
    a[9]  = (_Float16)(rsv * (a2.y + rsv * x2.y));
    a[10] = (_Float16)(rsv * (a2.z + rsv * x2.z));
    a[11] = (_Float16)(rsv * (a2.w + rsv * x2.w));
    a[12] = (_Float16)(rsv * (a3.x + rsv * x3.x));
    a[13] = (_Float16)(rsv * (a3.y + rsv * x3.y));
    a[14] = (_Float16)(rsv * (a3.z + rsv * x3.z));
    a[15] = (_Float16)(rsv * (a3.w + rsv * x3.w));
    c = __builtin_amdgcn_wmma_f32_16x16x32_f16(false, a, false, b,
                                               (short)0, c, false, false);
  }
  float bias = b1[col];
#pragma unroll
  for (int r = 0; r < 8; ++r) {
    int orow = mt * 16 + hi8 * 8 + r;   // D: VGPR r -> M = r + 8*hi
    if (orow < n_nodes) {
      float v = fmaxf(c[r] + bias, 0.0f);
      h[(size_t)orow * HID_SZ + col] = (_Float16)v;
    }
  }
}

// ---------------- WMMA GEMM layer 2 ----------------
// ps = rs[row] * (h @ w2)   (n x 64, f32) — pre-scaled for the 2nd aggregation
// grid.x = n_nodes/16; block = 128 = 4 waves; wave w -> N-tile w.
__global__ __launch_bounds__(128)
void gemm2_wmma(const _Float16* h, const v16h* w2s, const float* rs,
                float* ps, int n_nodes) {
  int wave = threadIdx.x >> 5;
  int lane = threadIdx.x & 31;
  int lh   = lane & 15;
  int hi8  = lane >> 4;
  int mt   = blockIdx.x;
  int nt   = wave;
  int row  = mt * 16 + lh;
  int arow = (row < n_nodes) ? row : 0;
  const _Float16* ph = h + (size_t)arow * HID_SZ;
  int col = nt * 16 + lh;

  v8f c = {0.f, 0.f, 0.f, 0.f, 0.f, 0.f, 0.f, 0.f};
#pragma unroll
  for (int kt = 0; kt < 4; ++kt) {
    int k0 = kt * 32 + hi8 * 8;
    v16h b = w2s[(nt * 4 + kt) * 32 + lane];
    // two 16B contiguous f16 vector loads
    v8h alo = *(const v8h*)(ph + k0);
    v8h ahi = *(const v8h*)(ph + k0 + 16);
    v16h a = __builtin_shufflevector(alo, ahi,
                                     0, 1, 2, 3, 4, 5, 6, 7,
                                     8, 9, 10, 11, 12, 13, 14, 15);
    c = __builtin_amdgcn_wmma_f32_16x16x32_f16(false, a, false, b,
                                               (short)0, c, false, false);
  }
#pragma unroll
  for (int r = 0; r < 8; ++r) {
    int orow = mt * 16 + hi8 * 8 + r;
    if (orow < n_nodes)
      ps[(size_t)orow * OUT_SZ + col] = rs[orow] * c[r];
  }
}

// out[j][f] = rs[j] * (acc2[j][f] + ps[j][f]) + b2[f]
__global__ void finalize(const float* acc2, const float* ps, const float* rs,
                         const float* b2, float* out, int n_nodes) {
  long long idx = (long long)blockIdx.x * blockDim.x + threadIdx.x;
  long long total = (long long)n_nodes * OUT_SZ;
  if (idx >= total) return;
  int j = (int)(idx >> 6);
  int f = (int)(idx & 63);
  out[idx] = rs[j] * (acc2[idx] + ps[idx]) + b2[f];
}

// ---------------- launch ----------------

extern "C" void kernel_launch(void* const* d_in, const int* in_sizes, int n_in,
                              void* d_out, int out_size, void* d_ws, size_t ws_size,
                              hipStream_t stream) {
  const float*     x   = (const float*)d_in[0];
  const long long* ei  = (const long long*)d_in[1];   // int64 (2, E)
  const float*     w1  = (const float*)d_in[2];
  const float*     b1  = (const float*)d_in[3];
  const float*     w2  = (const float*)d_in[4];
  const float*     b2  = (const float*)d_in[5];
  float*           out = (float*)d_out;

  const int n_nodes = in_sizes[0] / IN_SZ;
  const int n_edges = in_sizes[1] / 2;
  const long long* src = ei;
  const long long* dst = ei + n_edges;

  // carve workspace
  char* w = (char*)d_ws;
  auto take = [&](size_t bytes) {
    char* p = w;
    w += (bytes + 255) & ~(size_t)255;
    return p;
  };
  float*    deg  = (float*)take((size_t)n_nodes * 4);
  float*    rs   = (float*)take((size_t)n_nodes * 4);
  float*    acc1 = (float*)take((size_t)n_nodes * IN_SZ * 4);
  _Float16* h    = (_Float16*)take((size_t)n_nodes * HID_SZ * 2);
  float*    ps   = (float*)take((size_t)n_nodes * OUT_SZ * 4);
  float*    acc2 = (float*)take((size_t)n_nodes * OUT_SZ * 4);
  _Float16* w1s  = (_Float16*)take((size_t)IN_SZ * HID_SZ * 2);
  _Float16* w2s  = (_Float16*)take((size_t)HID_SZ * OUT_SZ * 2);

  // init (must happen every call: atomics accumulate into these)
  fill_f32<<<2048, 256, 0, stream>>>(deg, 1.0f, n_nodes);
  fill_f32<<<4096, 256, 0, stream>>>(acc1, 0.0f, (long long)n_nodes * IN_SZ);
  fill_f32<<<4096, 256, 0, stream>>>(acc2, 0.0f, (long long)n_nodes * OUT_SZ);
  // pre-swizzle weights into WMMA B-fragment order (f16)
  swizzle_w<<<(IN_SZ * HID_SZ + 255) / 256, 256, 0, stream>>>(w1, w1s, HID_SZ, 4);
  swizzle_w<<<(HID_SZ * OUT_SZ + 255) / 256, 256, 0, stream>>>(w2, w2s, OUT_SZ, 4);

  // degrees and rsqrt
  deg_count<<<2048, 256, 0, stream>>>(dst, deg, n_edges);
  rs_kernel<<<(n_nodes + 255) / 256, 256, 0, stream>>>(deg, rs, n_nodes);

  // layer 1: aggregate x, GEMM + relu
  int edge_blocks = (n_edges + 7) / 8;  // 8 waves (edges) per 256-thread block
  agg_edges_128<<<edge_blocks, 256, 0, stream>>>(src, dst, rs, x, acc1, n_edges);
  int mtiles = (n_nodes + 15) / 16;
  gemm1_wmma<<<mtiles, 256, 0, stream>>>(acc1, x, rs, (const v16h*)w1s, b1, h, n_nodes);

  // layer 2: GEMM first (agg(h)@w2 == agg(h@w2), halves gather width), then aggregate
  gemm2_wmma<<<mtiles, 128, 0, stream>>>(h, (const v16h*)w2s, rs, ps, n_nodes);
  agg_edges_64<<<edge_blocks, 256, 0, stream>>>(src, dst, ps, acc2, n_edges);

  // epilogue: self loop + final norm + bias
  long long total = (long long)n_nodes * OUT_SZ;
  finalize<<<(int)((total + 255) / 256), 256, 0, stream>>>(acc2, ps, rs, b2, out, n_nodes);
}
